// DecoderWindowAttention_30399778521555
// MI455X (gfx1250) — matrix operand order repaired and checked by reading, not compiled, not run on hardware
//
#include <hip/hip_runtime.h>
#include <stdint.h>

#define DIM 192
#define HEADS 6
#define NTOK 64
#define NWIN 512
#define QK_SCALE 0.17677669529663687f  // 1/sqrt(32)

typedef __attribute__((ext_vector_type(16))) __bf16 v16bf;
typedef __attribute__((ext_vector_type(8)))  float  v8f;

union BF16x16 { v16bf v; unsigned int u[8]; };
union F32x8   { v8f   v; float        f[8]; };

__device__ __forceinline__ unsigned short f2bf(float x) {
  unsigned int u = __float_as_uint(x);
  u += 0x7FFFu + ((u >> 16) & 1u);          // round-to-nearest-even
  return (unsigned short)(u >> 16);
}
__device__ __forceinline__ unsigned int pk2(float lo, float hi) {
  return (unsigned int)f2bf(lo) | ((unsigned int)f2bf(hi) << 16);
}
// 8 contiguous f32 at p0 -> frag elems 0..7, 8 contiguous at p1 -> elems 8..15
__device__ __forceinline__ v16bf cvt16(const float* p0, const float* p1) {
  BF16x16 r;
  const float4* a = (const float4*)p0;
  const float4* b = (const float4*)p1;
  float4 x = a[0], y = a[1], zc = b[0], wv = b[1];
  r.u[0] = pk2(x.x, x.y);   r.u[1] = pk2(x.z, x.w);
  r.u[2] = pk2(y.x, y.y);   r.u[3] = pk2(y.z, y.w);
  r.u[4] = pk2(zc.x, zc.y); r.u[5] = pk2(zc.z, zc.w);
  r.u[6] = pk2(wv.x, wv.y); r.u[7] = pk2(wv.z, wv.w);
  return r.v;
}
// 8 bf16 (16B aligned) at p0 -> elems 0..7, 8 bf16 at p1 -> elems 8..15
__device__ __forceinline__ v16bf lds16(const unsigned short* p0, const unsigned short* p1) {
  BF16x16 r;
  uint4 a = *(const uint4*)p0;
  uint4 b = *(const uint4*)p1;
  r.u[0] = a.x; r.u[1] = a.y; r.u[2] = a.z; r.u[3] = a.w;
  r.u[4] = b.x; r.u[5] = b.y; r.u[6] = b.z; r.u[7] = b.w;
  return r.v;
}
#define WMMA_BF16(A,B,C) \
  __builtin_amdgcn_wmma_f32_16x16x32_bf16(false,(A),false,(B),(short)0,(C),false,false)

// per-lane async copy: 16 bytes global -> LDS (tracked by ASYNCcnt)
__device__ __forceinline__ void async_cp16(unsigned ldsaddr, unsigned long long gaddr) {
  asm volatile("global_load_async_to_lds_b128 %0, %1, off"
               :: "v"(ldsaddr), "v"(gaddr) : "memory");
}
__device__ __forceinline__ void async_cp4(unsigned ldsaddr, unsigned long long gaddr) {
  asm volatile("global_load_async_to_lds_b32 %0, %1, off"
               :: "v"(ldsaddr), "v"(gaddr) : "memory");
}

__global__ __launch_bounds__(256)
void swin_window_attn(const float* __restrict__ Q, const float* __restrict__ K,
                      const float* __restrict__ V, const float* __restrict__ WMASK,
                      const unsigned char* __restrict__ AMASK,
                      const float* __restrict__ WQ, const float* __restrict__ BQ,
                      const float* __restrict__ WK, const float* __restrict__ BK,
                      const float* __restrict__ WV, const float* __restrict__ BV,
                      const float* __restrict__ WP, const float* __restrict__ BP,
                      const float* __restrict__ RPB, const int* __restrict__ RIDX,
                      float* __restrict__ OUT)
{
  constexpr int QS = 200;   // bf16 stride for QH/KH/X rows (pad vs 192)
  constexpr int VS = 72;    // bf16 stride for VHT rows (pad vs 64)
  constexpr int PS = 72;    // bf16 stride for per-wave P slots
  extern __shared__ char smraw[];
  unsigned short* QH  = (unsigned short*)smraw;            // 25600 B
  unsigned short* KH  = QH  + 64 * QS;                     // 25600 B
  unsigned short* VHT = KH  + 64 * QS;                     // 27648 B (V transposed [c][n])
  unsigned short* XB  = VHT + 192 * VS;                    // 25600 B
  unsigned short* PB  = XB  + 64 * QS;                     // 18432 B (8 waves * 16 * PS)
  float*          EXT  = (float*)(smraw + 122880);         // 16384 B window-mask slice
  unsigned char*  AMB  = (unsigned char*)(smraw + 139264); //  4096 B attn-mask slice
  float*          RPBS = (float*)(smraw + 143360);         //  5408 B rpb table

  const int bw   = blockIdx.x;
  const int lane = threadIdx.x & 31;
  const int w    = threadIdx.x >> 5;
  const int col  = lane & 15;
  const int hf   = lane >> 4;               // half: 0 = lanes 0-15, 1 = lanes 16-31
  const int widx = bw % NWIN;

  // ---- Phase 0: kick off async global->LDS staging of masks + rpb table ----
  {
    const int t = threadIdx.x;
    const unsigned extb = (unsigned)(size_t)(void*)EXT;
    const unsigned long long wsrc = (unsigned long long)(size_t)(WMASK + (size_t)widx * 4096);
    #pragma unroll
    for (int i = 0; i < 4; ++i) {                 // 16 KB window-mask slice
      const unsigned off = (unsigned)(t + i * 256) * 16u;
      async_cp16(extb + off, wsrc + off);
    }
    const unsigned ambb = (unsigned)(size_t)(void*)AMB;
    const unsigned long long asrc = (unsigned long long)(size_t)(AMASK + (size_t)bw * 4096);
    async_cp16(ambb + (unsigned)t * 16u, asrc + (unsigned long long)t * 16u);  // 4 KB bool mask
    const unsigned rb = (unsigned)(size_t)(void*)RPBS;
    const unsigned long long rsrc = (unsigned long long)(size_t)RPB;
    for (int idx = t; idx < (2 * 8 - 1) * (2 * 8 - 1) * HEADS; idx += 256)     // 1350 f32
      async_cp4(rb + (unsigned)idx * 4u, rsrc + (unsigned long long)idx * 4u);
  }

  // ---------------- Phase 1: Q/K/V projections (bf16 WMMA) ----------------
  for (int task = w; task < 12; task += 8) {
    const int tu  = __builtin_amdgcn_readfirstlane(task);  // wave-uniform -> scalar branches
    const int mat = tu >> 2;              // 0=q, 1=k, 2=v
    const int mt  = tu & 3;
    const float* src = (mat == 0 ? Q : (mat == 1 ? K : V)) + (size_t)bw * NTOK * DIM;
    const float* Wm  = (mat == 0 ? WQ : (mat == 1 ? WK : WV));
    const float* Bm  = (mat == 0 ? BQ : (mat == 1 ? BK : BV));
    const int m = mt * 16 + col;          // A row for this lane
    v16bf a[6];
    #pragma unroll
    for (int kt = 0; kt < 6; ++kt) {
      const float* ap = src + m * DIM + kt * 32 + hf * 8;
      a[kt] = cvt16(ap, ap + 16);
    }
    for (int nt = 0; nt < 12; ++nt) {
      const int c = nt * 16 + col;        // output channel = B column
      F32x8 acc;
      #pragma unroll
      for (int r = 0; r < 8; ++r) acc.f[r] = 0.f;
      #pragma unroll
      for (int kt = 0; kt < 6; ++kt) {
        const float* wp = Wm + c * DIM + kt * 32 + hf * 8;  // W^T == B col-major
        v16bf bfrag = cvt16(wp, wp + 16);
        acc.v = WMMA_BF16(a[kt], bfrag, acc.v);
      }
      const float bias = Bm[c];
      if (mat == 0) {
        #pragma unroll
        for (int r = 0; r < 8; ++r)
          QH[(mt * 16 + hf * 8 + r) * QS + c] = f2bf((acc.f[r] + bias) * QK_SCALE);
      } else if (mat == 1) {
        #pragma unroll
        for (int r = 0; r < 8; ++r)
          KH[(mt * 16 + hf * 8 + r) * QS + c] = f2bf(acc.f[r] + bias);
      } else {
        #pragma unroll
        for (int r = 0; r < 8; ++r)
          VHT[c * VS + (mt * 16 + hf * 8 + r)] = f2bf(acc.f[r] + bias);  // transposed
      }
    }
  }
  asm volatile("s_wait_asynccnt 0x0" ::: "memory");   // staging done (this wave)
  __syncthreads();                                    // all waves: proj + staging visible

  // ------------- Phase 2: per-(head, row-tile) attention -------------
  for (int task = w; task < 24; task += 8) {
    const int h  = task >> 2;
    const int mt = task & 3;
    const unsigned short* qp = QH + (mt * 16 + col) * QS + h * 32 + hf * 8;
    v16bf qa = lds16(qp, qp + 16);
    F32x8 s[4];
    #pragma unroll
    for (int nt = 0; nt < 4; ++nt) {
      const unsigned short* kp = KH + (nt * 16 + col) * QS + h * 32 + hf * 8;
      v16bf kb = lds16(kp, kp + 16);
      F32x8 z;
      #pragma unroll
      for (int r = 0; r < 8; ++r) z.f[r] = 0.f;
      s[nt].v = WMMA_BF16(qa, kb, z.v);      // hd==32 -> single K-step
    }
    // masks + relative-position bias (masks/table now in LDS)
    #pragma unroll
    for (int nt = 0; nt < 4; ++nt) {
      const int n = nt * 16 + col;
      #pragma unroll
      for (int r = 0; r < 8; ++r) {
        const int m  = mt * 16 + hf * 8 + r;
        const int mn = m * 64 + n;
        float val = AMB[mn] ? -1e9f : s[nt].f[r];
        const int ri = RIDX[mn];                      // L2-resident global gather
        val += RPBS[ri * HEADS + h] + EXT[mn];
        s[nt].f[r] = val;
      }
    }
    // softmax over the 64 keys: 4 frags x 16 lanes of this half hold one row
    #pragma unroll
    for (int r = 0; r < 8; ++r) {
      float mx = fmaxf(fmaxf(s[0].f[r], s[1].f[r]), fmaxf(s[2].f[r], s[3].f[r]));
      mx = fmaxf(mx, __shfl_xor(mx, 1, 32));
      mx = fmaxf(mx, __shfl_xor(mx, 2, 32));
      mx = fmaxf(mx, __shfl_xor(mx, 4, 32));
      mx = fmaxf(mx, __shfl_xor(mx, 8, 32));
      float e0 = __expf(s[0].f[r] - mx);
      float e1 = __expf(s[1].f[r] - mx);
      float e2 = __expf(s[2].f[r] - mx);
      float e3 = __expf(s[3].f[r] - mx);
      float sum = e0 + e1 + e2 + e3;
      sum += __shfl_xor(sum, 1, 32);
      sum += __shfl_xor(sum, 2, 32);
      sum += __shfl_xor(sum, 4, 32);
      sum += __shfl_xor(sum, 8, 32);
      const float inv = 1.0f / sum;
      s[0].f[r] = e0 * inv; s[1].f[r] = e1 * inv;
      s[2].f[r] = e2 * inv; s[3].f[r] = e3 * inv;
    }
    // C-layout -> A-layout via this wave's private LDS slot (DS in-order per wave)
    unsigned short* Pslot = PB + w * 16 * PS;
    #pragma unroll
    for (int nt = 0; nt < 4; ++nt) {
      const int n = nt * 16 + col;
      #pragma unroll
      for (int r = 0; r < 8; ++r)
        Pslot[(hf * 8 + r) * PS + n] = f2bf(s[nt].f[r]);
    }
    // O(16x32) = P(16x64) @ Vh(64x32)
    F32x8 o[2];
    #pragma unroll
    for (int r = 0; r < 8; ++r) { o[0].f[r] = 0.f; o[1].f[r] = 0.f; }
    #pragma unroll
    for (int kt = 0; kt < 2; ++kt) {
      const unsigned short* pp = Pslot + col * PS + kt * 32 + hf * 8;
      v16bf pa = lds16(pp, pp + 16);
      #pragma unroll
      for (int nt2 = 0; nt2 < 2; ++nt2) {
        const unsigned short* vp = VHT + (h * 32 + nt2 * 16 + col) * VS + kt * 32 + hf * 8;
        v16bf vb = lds16(vp, vp + 16);
        o[nt2].v = WMMA_BF16(pa, vb, o[nt2].v);
      }
    }
    #pragma unroll
    for (int nt2 = 0; nt2 < 2; ++nt2)
      #pragma unroll
      for (int r = 0; r < 8; ++r)
        XB[(mt * 16 + hf * 8 + r) * QS + h * 32 + nt2 * 16 + col] = f2bf(o[nt2].f[r]);
  }
  __syncthreads();

  // ---------------- Phase 3: output projection ----------------
  for (int task = w; task < 48; task += 8) {
    const int tu = __builtin_amdgcn_readfirstlane(task);
    const int mt = tu / 12;
    const int nt = tu % 12;
    const int c  = nt * 16 + col;
    F32x8 acc;
    #pragma unroll
    for (int r = 0; r < 8; ++r) acc.f[r] = 0.f;
    #pragma unroll
    for (int kt = 0; kt < 6; ++kt) {
      const unsigned short* xp = XB + (mt * 16 + col) * QS + kt * 32 + hf * 8;
      v16bf xa = lds16(xp, xp + 16);
      const float* wp = WP + c * DIM + kt * 32 + hf * 8;
      v16bf wb = cvt16(wp, wp + 16);
      acc.v = WMMA_BF16(xa, wb, acc.v);
    }
    const float bias = BP[c];
    #pragma unroll
    for (int r = 0; r < 8; ++r)
      OUT[(size_t)bw * NTOK * DIM + (mt * 16 + hf * 8 + r) * DIM + c] = acc.f[r] + bias;
  }
}

extern "C" void kernel_launch(void* const* d_in, const int* in_sizes, int n_in,
                              void* d_out, int out_size, void* d_ws, size_t ws_size,
                              hipStream_t stream) {
  (void)n_in; (void)out_size; (void)d_ws; (void)ws_size;
  const float*         Q     = (const float*)d_in[0];
  const float*         K     = (const float*)d_in[1];
  const float*         V     = (const float*)d_in[2];
  const float*         WMASK = (const float*)d_in[3];
  const unsigned char* AMASK = (const unsigned char*)d_in[4];  // bool
  const float*         WQ    = (const float*)d_in[5];
  const float*         BQ    = (const float*)d_in[6];
  const float*         WK    = (const float*)d_in[7];
  const float*         BK    = (const float*)d_in[8];
  const float*         WV    = (const float*)d_in[9];
  const float*         BV    = (const float*)d_in[10];
  const float*         WP    = (const float*)d_in[11];
  const float*         BP    = (const float*)d_in[12];
  const float*         RPB   = (const float*)d_in[13];
  const int*           RIDX  = (const int*)d_in[14];
  float* OUT = (float*)d_out;

  const int Bw = in_sizes[0] / (NTOK * DIM);   // 2048 windows
  const size_t shmem = 148768;                 // QH/KH/VHT/XB/PB + EXT + AMB + RPBS
  swin_window_attn<<<Bw, 256, shmem, stream>>>(Q, K, V, WMASK, AMASK,
                                               WQ, BQ, WK, BK, WV, BV, WP, BP,
                                               RPB, RIDX, OUT);
}
